// ScaledDotProductAttention_5248450036070
// MI455X (gfx1250) — matrix operand order repaired and checked by reading, not compiled, not run on hardware
//
#include <hip/hip_runtime.h>
#include <cstdint>
#include <cstddef>

typedef __attribute__((ext_vector_type(16))) __bf16 v16bf;
typedef __attribute__((ext_vector_type(8)))  float  v8f;

#define BATCH   8
#define SEQ     2048
#define DIM     1024
#define BR      32          // query rows per workgroup (2 strips of 16)
#define BC      32          // keys per tile
#define NSTRIP  2           // query strips per wave -> K fragment reuse x2
#define NWAVES  8
#define WSLICE  128         // DIM / NWAVES, per-wave D slice
#define VT_STRIDE 40        // BC + 8 pad (bf16) -> 80B rows, 16B aligned
#define P_STRIDE  48        // bf16 units -> 96B rows, fragment chunks 16B aligned
#define SCALE   0.03125f    // 1/sqrt(1024), folded into Q conversion

static __device__ __forceinline__ uint32_t pk_bf16(float a, float b) {
    uint32_t lo = __builtin_bit_cast(unsigned short, (__bf16)a);
    uint32_t hi = __builtin_bit_cast(unsigned short, (__bf16)b);
    return (hi << 16) | lo;
}

__global__ __launch_bounds__(256, 4)
void fa_fwd_kernel(const float* __restrict__ Qg,
                   const float* __restrict__ Kg,
                   const float* __restrict__ Vg,
                   float* __restrict__ Og)
{
    __shared__ __attribute__((aligned(16))) __bf16 Vt[DIM][VT_STRIDE];          // V tile transposed [d][k]
    __shared__ __attribute__((aligned(16))) float  Sred[2][BR][BC];             // double-buffered reduction
    __shared__ __attribute__((aligned(16))) __bf16 Pbuf[NWAVES][BR][P_STRIDE];  // per-wave P (bf16)
    __shared__ __attribute__((aligned(16))) float  alphaBuf[NWAVES][BR];        // per-wave rescale bcast

    const int tid  = threadIdx.x;
    const int lane = tid & 31;
    const int wave = tid >> 5;
    const int half = lane >> 4;
    const int lm   = lane & 15;

    const int qt    = blockIdx.x;
    const int b     = blockIdx.y;
    const int qbase = qt * BR;
    const int dbase = wave * WSLICE;
    const size_t bo = (size_t)b * SEQ * DIM;

    // V staging assignment: row-pair + 64-wide d chunk per thread (packed b32 stores)
    const int rp  = tid >> 4;          // 0..15 -> key rows {2rp, 2rp+1}
    const int dv0 = (tid & 15) * 64;

    // ---- preload Q A-fragments for both strips, pre-scaled by 1/sqrt(D) ----
    v16bf qf[NSTRIP][4];
    {
        const int off = half ? 16 : 0;
#pragma unroll
        for (int s = 0; s < NSTRIP; ++s) {
            const float* qp = Qg + bo + (size_t)(qbase + s * 16 + lm) * DIM + dbase;
#pragma unroll
            for (int kb = 0; kb < 4; ++kb) {
                const float* p0 = qp + kb * 32 + off;
                const float* p1 = p0 + 8;
#pragma unroll
                for (int e = 0; e < 8; ++e) {
                    qf[s][kb][e]     = (__bf16)(p0[e] * SCALE);
                    qf[s][kb][8 + e] = (__bf16)(p1[e] * SCALE);
                }
            }
        }
    }

    v8f oacc[NSTRIP][8];
#pragma unroll
    for (int s = 0; s < NSTRIP; ++s)
#pragma unroll
        for (int nb = 0; nb < 8; ++nb) oacc[s][nb] = v8f{};

    // online-softmax running state: lane l tracks query row (qbase + l); identical in all waves
    float mrun = -1e30f, lrun = 0.0f;

    const int ntiles = qt + 1;      // BR == BC == 32: causal => exactly qt+1 key tiles

    // pre-zero both Sred buffers (visible after first barrier)
#pragma unroll
    for (int z = 0; z < 8; ++z) ((float*)Sred)[tid + z * 256] = 0.0f;

    for (int kt = 0; kt < ntiles; ++kt) {
        const int ktbase = kt * BC;
        const int p      = kt & 1;

        // B1: PV(kt-1) done (Vt free), softmax(kt-1) reads of Sred[p^1] done,
        //     zeroing of Sred[p] (done in iter kt-1) visible.
        __syncthreads();

        // ================= phase 1: stage Vt(kt) overlapped with QK(kt) =================

        // ---- stage V tile transposed -> LDS bf16 (packed b32 stores) ----
        {
            const float* vp0 = Vg + bo + (size_t)(ktbase + 2 * rp) * DIM + dv0;
            const float* vp1 = vp0 + DIM;
#pragma unroll 4
            for (int c = 0; c < 64; c += 4) {
                float4 f0 = *(const float4*)(vp0 + c);
                float4 f1 = *(const float4*)(vp1 + c);
                *(uint32_t*)&Vt[dv0 + c + 0][2 * rp] = pk_bf16(f0.x, f1.x);
                *(uint32_t*)&Vt[dv0 + c + 1][2 * rp] = pk_bf16(f0.y, f1.y);
                *(uint32_t*)&Vt[dv0 + c + 2][2 * rp] = pk_bf16(f0.z, f1.z);
                *(uint32_t*)&Vt[dv0 + c + 3][2 * rp] = pk_bf16(f0.w, f1.w);
            }
        }
        // zero the other Sred buffer for iteration kt+1
        {
            float* sz = &Sred[p ^ 1][0][0];
#pragma unroll
            for (int z = 0; z < 4; ++z) sz[tid + z * 256] = 0.0f;
        }

        // ---- prefetch next tile's K fragments and V rows ----
        {
            const int ktn = (kt + 1 < ntiles) ? (kt + 1) : kt;
            const size_t knext = bo + (size_t)(ktn * BC) * DIM;
#pragma unroll
            for (int cb = 0; cb < 2; ++cb)
#pragma unroll
                for (int kb = 0; kb < 4; ++kb)
                    __builtin_prefetch(Kg + knext + (size_t)(cb * 16 + lm) * DIM
                                       + dbase + kb * 32 + (half ? 16 : 0), 0, 1);
            __builtin_prefetch(Vg + knext + (size_t)(2 * rp) * DIM + dv0, 0, 1);
            __builtin_prefetch(Vg + knext + (size_t)(2 * rp + 1) * DIM + dv0, 0, 1);
        }

        // ---- QK^T partials over this wave's D slice; each K fragment feeds BOTH strips ----
#pragma unroll
        for (int cb = 0; cb < 2; ++cb) {
            v8f sacc0 = v8f{}, sacc1 = v8f{};
#pragma unroll
            for (int kb = 0; kb < 4; ++kb) {
                const float* kp = Kg + bo + (size_t)(ktbase + cb * 16 + lm) * DIM
                                + dbase + kb * 32 + (half ? 16 : 0);
                v16bf kf;
#pragma unroll
                for (int e = 0; e < 16; ++e) kf[e] = (__bf16)kp[e];
                sacc0 = __builtin_amdgcn_wmma_f32_16x16x32_bf16(
                    false, qf[0][kb], false, kf, (short)0, sacc0, false, false);
                sacc1 = __builtin_amdgcn_wmma_f32_16x16x32_bf16(
                    false, qf[1][kb], false, kf, (short)0, sacc1, false, false);
            }
#pragma unroll
            for (int i = 0; i < 8; ++i) {
                atomicAdd(&Sred[p][i + half * 8][cb * 16 + lm], sacc0[i]);
                atomicAdd(&Sred[p][16 + i + half * 8][cb * 16 + lm], sacc1[i]);
            }
        }

        // B2: staging + reduction complete
        __syncthreads();

        // ================= phase 2: softmax + PV =================

        // ---- online softmax, computed redundantly per wave (lane = row), no barrier ----
        {
            const int r = lane;                 // row 0..31 (both strips)
            float mx = -1e30f;
#pragma unroll
            for (int c = 0; c < BC; ++c) {
                float v = (ktbase + c > qbase + r) ? -1e30f : Sred[p][r][c];
                mx = fmaxf(mx, v);
            }
            const float mnew  = fmaxf(mrun, mx);
            const float alpha = __expf(mrun - mnew);
            float sum = 0.0f;
#pragma unroll
            for (int c = 0; c < BC; ++c) {
                float v = (ktbase + c > qbase + r) ? -1e30f : Sred[p][r][c];
                float pr = __expf(v - mnew);
                sum += pr;
                Pbuf[wave][r][c] = (__bf16)pr;
            }
            lrun = alpha * lrun + sum;
            mrun = mnew;
            alphaBuf[wave][r] = alpha;
        }
        // in-wave DS ordering makes Pbuf/alphaBuf visible to this wave without a barrier

        // ---- per strip: rescale O accumulators, O += P * V_slice ----
#pragma unroll
        for (int s = 0; s < NSTRIP; ++s) {
            float al[8];
#pragma unroll
            for (int i = 0; i < 8; ++i) al[i] = alphaBuf[wave][s * 16 + half * 8 + i];
#pragma unroll
            for (int nb = 0; nb < 8; ++nb)
#pragma unroll
                for (int i = 0; i < 8; ++i) oacc[s][nb][i] *= al[i];

            v16bf pf;
            {
                const int off = half ? 16 : 0;
                __builtin_memcpy(&pf, &Pbuf[wave][s * 16 + lm][off], 16);
                __builtin_memcpy((char*)&pf + 16, &Pbuf[wave][s * 16 + lm][8 + off], 16);
            }
#pragma unroll
            for (int nb = 0; nb < 8; ++nb) {
                const int dv = dbase + nb * 16 + lm;
                v16bf vf;
                __builtin_memcpy(&vf, &Vt[dv][half * 16], 32);
                oacc[s][nb] = __builtin_amdgcn_wmma_f32_16x16x32_bf16(
                    false, pf, false, vf, (short)0, oacc[s][nb], false, false);
            }
        }
    }

    // ---- finalize: divide by softmax denominator (per-wave broadcast), store fp32 ----
    alphaBuf[wave][lane] = 1.0f / lrun;    // own buffer, in-wave ordering
#pragma unroll
    for (int s = 0; s < NSTRIP; ++s) {
        float linv[8];
#pragma unroll
        for (int i = 0; i < 8; ++i) linv[i] = alphaBuf[wave][s * 16 + half * 8 + i];
#pragma unroll
        for (int nb = 0; nb < 8; ++nb)
#pragma unroll
            for (int i = 0; i < 8; ++i)
                Og[bo + (size_t)(qbase + s * 16 + i + half * 8) * DIM + dbase + nb * 16 + lm]
                    = oacc[s][nb][i] * linv[i];
    }
}

extern "C" void kernel_launch(void* const* d_in, const int* in_sizes, int n_in,
                              void* d_out, int out_size, void* d_ws, size_t ws_size,
                              hipStream_t stream)
{
    (void)in_sizes; (void)n_in; (void)out_size; (void)d_ws; (void)ws_size;
    const float* Q = (const float*)d_in[0];
    const float* K = (const float*)d_in[1];
    const float* V = (const float*)d_in[2];
    // d_in[3] is the boolean causal mask; causality is computed analytically.
    float* O = (float*)d_out;

    dim3 grid(SEQ / BR, BATCH);
    dim3 block(256);
    fa_fwd_kernel<<<grid, block, 0, stream>>>(Q, K, V, O);
}
